// ActorCriticListenerConvAblationQuantizeReady_90941637525572
// MI455X (gfx1250) — matrix operand order repaired and checked by reading, not compiled, not run on hardware
//
#include <hip/hip_runtime.h>
#include <hip/hip_bf16.h>

typedef __bf16 bf16_t;
typedef __attribute__((ext_vector_type(16))) __bf16 v16bf;
typedef __attribute__((ext_vector_type(8)))  __bf16 v8bf;
typedef __attribute__((ext_vector_type(8)))  float  v8f;
typedef int v4i __attribute__((vector_size(4 * sizeof(int))));

#define WMMA_BF16(a, b, c) \
  __builtin_amdgcn_wmma_f32_16x16x32_bf16(false, (a), false, (b), (short)0, (c), false, false)

#define AS1 __attribute__((address_space(1)))
#define AS3 __attribute__((address_space(3)))

#if defined(__HIP_DEVICE_COMPILE__) && \
    __has_builtin(__builtin_amdgcn_global_load_async_to_lds_b128) && \
    __has_builtin(__builtin_amdgcn_s_wait_asynccnt)
#define HAVE_ASYNC_LDS 1
#else
#define HAVE_ASYNC_LDS 0
#endif

static constexpr int PIX = 512 * 32 * 32;   // 524288 pixels

// ---------------- fragment-order (swizzled) index helpers ----------------
// A-fragment lane layout for 16x16x32 bf16 WMMA:
//   lane = hi*16 + m ; e0..7 -> k = 8*hi + e ; e8..15 -> k = 16 + 8*hi + (e-8)
__device__ __forceinline__ int a_sw_idx(int m, int k) {
  int kc = k >> 5, r = k & 31;
  int hi = (r >> 3) & 1;
  int e  = (r & 7) + (((r >> 4) & 1) << 3);
  return ((kc << 5) + (hi << 4) + m) * 16 + e;
}

// B-fragment lane layout: lane = hi*16 + n ; e0..7 -> k = 8*hi + e ; e8..15 -> k = 16 + 8*hi + (e-8)
__device__ __forceinline__ int b_sw_idx(int k, int n, int NT) {
  int kc = k >> 5, r = k & 31;
  int hi = (r >> 3) & 1, e = (r & 7) + (((r >> 4) & 1) << 3);
  int nt = n >> 4, nl = n & 15;
  return (((kc * NT + nt) << 5) + (hi << 4) + nl) * 16 + e;
}

__device__ __forceinline__ v16bf load_frag16(const bf16_t* p) {
  return *(const v16bf*)p;   // 32B contiguous -> 2x ds_load_b128 / global b128
}

__device__ __forceinline__ v16bf combine8(v8bf lo, v8bf hi) {
  return __builtin_shufflevector(lo, hi, 0, 1, 2, 3, 4, 5, 6, 7,
                                         8, 9, 10, 11, 12, 13, 14, 15);
}

// ---------------- init: zero the loss accumulator ----------------
__global__ void k_init(float* loss) {
  if (threadIdx.x == 0 && blockIdx.x == 0) loss[0] = 0.0f;
}

// ---------------- conv1: 1 -> 32 channels, 3x3 SAME, ReLU ----------------
__global__ __launch_bounds__(256) void k_conv1(const float* __restrict__ x,
                                               const float* __restrict__ w,
                                               const float* __restrict__ b,
                                               bf16_t* __restrict__ h1) {
  __shared__ float wl[288];
  __shared__ float bl[32];
  int tid = threadIdx.x;
  if (tid < 288) wl[tid] = w[tid];
  if (tid < 32)  bl[tid] = b[tid];
  __syncthreads();

  int pix = blockIdx.x * 256 + tid;
  int img = pix >> 10, p = pix & 1023;
  int y = p >> 5, xc = p & 31;

  float acc[32];
#pragma unroll
  for (int c = 0; c < 32; ++c) acc[c] = 0.0f;

  for (int t = 0; t < 9; ++t) {
    int dy = t / 3 - 1, dx = t % 3 - 1;
    int yy = y + dy, xx = xc + dx;
    if (yy < 0 || yy > 31 || xx < 0 || xx > 31) continue;
    float xv = x[(img << 10) + (yy << 5) + xx];
#pragma unroll
    for (int c = 0; c < 32; ++c) acc[c] = fmaf(xv, wl[t * 32 + c], acc[c]);
  }
  bf16_t* o = h1 + (size_t)pix * 32;
#pragma unroll
  for (int g = 0; g < 4; ++g) {
    v8bf pk;
#pragma unroll
    for (int j = 0; j < 8; ++j)
      pk[j] = (bf16_t)fmaxf(acc[g * 8 + j] + bl[g * 8 + j], 0.0f);
    *(v8bf*)(o + g * 8) = pk;
  }
}

// ---------------- conv2: implicit GEMM [pix x 288] x [288 x 64], WMMA ----------------
__global__ __launch_bounds__(128) void k_conv2(const bf16_t* __restrict__ h1,
                                               const float* __restrict__ w2,
                                               const float* __restrict__ b2,
                                               bf16_t* __restrict__ h2) {
  __shared__ __align__(32) bf16_t wl[288 * 64];   // 36 KB, B-fragment order
  int tid = threadIdx.x;
  for (int i = tid; i < 288 * 64; i += 128) {
    int kall = i >> 6;          // tap*32 + cin
    int n    = i & 63;          // cout
    int tap  = kall >> 5;
    int r    = kall & 31;
    int hi   = r >> 4, e = r & 15;
    int nt   = n >> 4, nl = n & 15;
    wl[((((tap << 2) + nt) << 5) + (hi << 4) + nl) * 16 + e] = (bf16_t)w2[i];
  }
  __syncthreads();

  int img = blockIdx.x >> 4;
  int rp = blockIdx.x & 15;
  int y0 = rp * 2;
  int wave = tid >> 5, lane = tid & 31;
  int m = lane & 15, hi = lane >> 4;
  int lp = wave * 16 + m;
  int y = y0 + (lp >> 5), xc = lp & 31;

  v8f acc0 = {}, acc1 = {}, acc2 = {}, acc3 = {};
  for (int t = 0; t < 9; ++t) {
    int dy = t / 3 - 1, dx = t % 3 - 1;
    int yy = y + dy, xx = xc + dx;
    v16bf a;
    if (yy >= 0 && yy < 32 && xx >= 0 && xx < 32) {
      const bf16_t* p = h1 + ((size_t)((img << 10) + (yy << 5) + xx)) * 32;
      v8bf lo = *(const v8bf*)(p + 16 * hi);
      v8bf hv = *(const v8bf*)(p + 8 + 16 * hi);
      a = combine8(lo, hv);
    } else {
#pragma unroll
      for (int e = 0; e < 16; ++e) a[e] = (bf16_t)0.0f;
    }
    const bf16_t* bb = wl + ((t << 2) << 5) * 16 + lane * 16;
    v16bf b0 = load_frag16(bb);
    v16bf b1 = load_frag16(bb + 32 * 16);
    v16bf b2f = load_frag16(bb + 64 * 16);
    v16bf b3 = load_frag16(bb + 96 * 16);
    acc0 = WMMA_BF16(a, b0, acc0);
    acc1 = WMMA_BF16(a, b1, acc1);
    acc2 = WMMA_BF16(a, b2f, acc2);
    acc3 = WMMA_BF16(a, b3, acc3);
  }
  size_t pixbase = (size_t)blockIdx.x * 64 + wave * 16;
  float bv0 = b2[m], bv1 = b2[m + 16], bv2 = b2[m + 32], bv3 = b2[m + 48];
#pragma unroll
  for (int r = 0; r < 8; ++r) {
    int row = r + 8 * hi;
    bf16_t* o = h2 + (pixbase + row) * 64 + m;
    o[0]  = (bf16_t)fmaxf(acc0[r] + bv0, 0.0f);
    o[16] = (bf16_t)fmaxf(acc1[r] + bv1, 0.0f);
    o[32] = (bf16_t)fmaxf(acc2[r] + bv2, 0.0f);
    o[48] = (bf16_t)fmaxf(acc3[r] + bv3, 0.0f);
  }
}

// ---------------- VQ: distances [pix x 64] x [64 x 512] + fused argmin ----------------
__global__ __launch_bounds__(256) void k_vq(const bf16_t* __restrict__ h2,
                                            const float* __restrict__ codebook,
                                            int* __restrict__ idx) {
  __shared__ __align__(32) bf16_t cb[256 * 64];   // 32 KB, code-major
  __shared__ float cn[256];
  int tid = threadIdx.x;
  int wave = tid >> 5, lane = tid & 31, m = lane & 15, hi = lane >> 4;

  size_t pix = (size_t)blockIdx.x * 128 + wave * 16 + m;
  const bf16_t* rp = h2 + pix * 64;
  v16bf a0 = combine8(*(const v8bf*)(rp + 16 * hi), *(const v8bf*)(rp + 8 + 16 * hi));
  v16bf a1 = combine8(*(const v8bf*)(rp + 32 + 16 * hi), *(const v8bf*)(rp + 40 + 16 * hi));

  float bestv[8];
  int besti[8];
#pragma unroll
  for (int r = 0; r < 8; ++r) { bestv[r] = 3.4e38f; besti[r] = 0; }

  for (int ph = 0; ph < 2; ++ph) {
    __syncthreads();
    for (int i = tid; i < 256 * 64; i += 256) cb[i] = (bf16_t)codebook[ph * 16384 + i];
    {
      float s = 0.0f;
      const float* cr = codebook + (size_t)(ph * 256 + tid) * 64;
      for (int d = 0; d < 64; ++d) { float v = cr[d]; s = fmaf(v, v, s); }
      cn[tid] = s;
    }
    __syncthreads();

    for (int nt = 0; nt < 16; ++nt) {
      const bf16_t* crow = cb + (size_t)(nt * 16 + m) * 64;
      v16bf b0 = load_frag16(crow + 16 * hi);
      v16bf b1 = load_frag16(crow + 32 + 16 * hi);
      v8f acc = {};
      acc = WMMA_BF16(a0, b0, acc);
      acc = WMMA_BF16(a1, b1, acc);
      float cnv = cn[nt * 16 + m];
      int nglob = ph * 256 + nt * 16 + m;
#pragma unroll
      for (int r = 0; r < 8; ++r) {
        float d = cnv - 2.0f * acc[r];
        if (d < bestv[r]) { bestv[r] = d; besti[r] = nglob; }
      }
    }
  }
#pragma unroll
  for (int r = 0; r < 8; ++r) {
    float v = bestv[r];
    int bi = besti[r];
    for (int off = 8; off >= 1; off >>= 1) {
      float ov = __shfl_xor(v, off);
      int oi = __shfl_xor(bi, off);
      if (ov < v || (ov == v && oi < bi)) { v = ov; bi = oi; }
    }
    bestv[r] = v; besti[r] = bi;
  }
  if (m == 0) {
#pragma unroll
    for (int r = 0; r < 8; ++r) {
      size_t prow = (size_t)blockIdx.x * 128 + wave * 16 + r + 8 * hi;
      idx[prow] = besti[r];
    }
  }
}

// ---------------- gather quantized, vq-loss partial sums, emit float indices ----------------
__global__ __launch_bounds__(256) void k_gather(const int* __restrict__ idx,
                                                const float* __restrict__ codebook,
                                                const bf16_t* __restrict__ h2,
                                                bf16_t* __restrict__ emb,
                                                float* __restrict__ loss,
                                                float* __restrict__ out_idx) {
  size_t n = (size_t)blockIdx.x * 256 + threadIdx.x;
  int id = idx[n];
  out_idx[n] = (float)id;
  const float* cr = codebook + (size_t)id * 64;
  const bf16_t* hr = h2 + n * 64;
  bf16_t* er = emb + n * 64;
  float s = 0.0f;
#pragma unroll
  for (int g = 0; g < 8; ++g) {
    v8bf hv = *(const v8bf*)(hr + g * 8);
    v8bf qv;
#pragma unroll
    for (int j = 0; j < 8; ++j) {
      float q = cr[g * 8 + j];
      float h = (float)hv[j];
      float df = q - h;
      s = fmaf(df, df, s);
      qv[j] = (bf16_t)q;
    }
    *(v8bf*)(er + g * 8) = qv;
  }
  for (int off = 16; off >= 1; off >>= 1) s += __shfl_xor(s, off);
  if ((threadIdx.x & 31) == 0) atomicAdd(loss, s);
}

// ---------------- fc1: [512 x 65536] x [65536 x 128], WMMA, K-chunked ----------------
__global__ __launch_bounds__(256) void k_fc1(const bf16_t* __restrict__ emb,
                                             const float* __restrict__ w1,
                                             const float* __restrict__ b1,
                                             float* __restrict__ emb1) {
  __shared__ __align__(32) bf16_t As[16 * 128];    // 4 KB, A-fragment order
  __shared__ __align__(32) bf16_t Bs[128 * 128];   // 32 KB, B-fragment order
  int tid = threadIdx.x, wave = tid >> 5, lane = tid & 31;
  int m = lane & 15, hi = lane >> 4;
  int mbase = blockIdx.x * 16, n0 = wave * 16;

  // A-staging decomposition (constant per thread): 16B run -> 16B run
  int arow = tid >> 4, ag = tid & 15;
  int akc = ag >> 2, asub = ag & 3;
  int ahi = asub >> 1, ae0 = (asub & 1) << 3;
  bf16_t* adst = As + ((akc << 5) + (ahi << 4) + arow) * 16 + ae0;
  // B-staging decomposition (constant per thread)
  int bk = tid >> 1, bn0 = (tid & 1) << 6;
  int bkc = bk >> 5, br = bk & 31;
  int bhi = br >> 4, be = br & 15;

  v8f acc = {};
  for (int kk = 0; kk < 65536; kk += 128) {
    __syncthreads();
    {   // stage A: async global->LDS b128 when available, else VGPR copy
      const bf16_t* src = emb + (size_t)(mbase + arow) * 65536 + kk + ag * 8;
#if HAVE_ASYNC_LDS
      __builtin_amdgcn_global_load_async_to_lds_b128(
          (AS1 v4i*)src, (AS3 v4i*)adst, 0, 0);
#else
#pragma unroll
      for (int j = 0; j < 8; ++j) adst[j] = src[j];
#endif
    }
    {   // stage B (f32 -> bf16, coalesced source, swizzled dest)
      const float* src = w1 + (size_t)(kk + bk) * 128 + bn0;
#pragma unroll
      for (int j = 0; j < 64; ++j) {
        int n = bn0 + j;
        Bs[((((bkc << 3) + (n >> 4)) << 5) + (bhi << 4) + (n & 15)) * 16 + be] =
            (bf16_t)src[j];
      }
      if (kk + 128 < 65536) __builtin_prefetch(src + 128 * 128, 0, 1);
    }
#if HAVE_ASYNC_LDS
    __builtin_amdgcn_s_wait_asynccnt(0);
#endif
    __syncthreads();
#pragma unroll
    for (int kc = 0; kc < 4; ++kc) {
      v16bf a = load_frag16(As + ((kc << 5) + lane) * 16);
      v16bf b = load_frag16(Bs + ((((kc << 3) + wave) << 5) + lane) * 16);
      acc = WMMA_BF16(a, b, acc);
    }
  }
#pragma unroll
  for (int r = 0; r < 8; ++r) {
    int row = r + 8 * hi;
    float v = acc[r] + b1[n0 + m];
    emb1[(size_t)(mbase + row) * 128 + n0 + m] = fmaxf(v, 0.0f);
  }
}

// ---------------- fused 128x128 dense layer in LDS (8 waves x 16 cols) ----------------
__device__ void dense128(const float* __restrict__ W, const float* __restrict__ bias,
                         const bf16_t* in, bf16_t* out,
                         bf16_t* wl, float* bl, int tid) {
  for (int i = tid; i < 128 * 128; i += 256) {
    int k = i >> 7, n = i & 127;
    wl[b_sw_idx(k, n, 8)] = (bf16_t)W[i];
  }
  if (tid < 128) bl[tid] = bias[tid];
  __syncthreads();
  int wave = tid >> 5, lane = tid & 31, m = lane & 15, hi = lane >> 4, n0 = wave * 16;
  v8f acc = {};
#pragma unroll
  for (int kc = 0; kc < 4; ++kc) {
    v16bf a = load_frag16(in + ((kc << 5) + lane) * 16);
    v16bf b = load_frag16(wl + ((((kc << 3) + wave) << 5) + lane) * 16);
    acc = WMMA_BF16(a, b, acc);
  }
  __syncthreads();   // allow in == out
#pragma unroll
  for (int r = 0; r < 8; ++r) {
    int row = r + 8 * hi;
    float v = acc[r] + bl[n0 + m];
    out[a_sw_idx(row, n0 + m)] = (bf16_t)fmaxf(v, 0.0f);
  }
  __syncthreads();
}

// ---------------- trunk + actor/critic heads + softmax + loss writeout ----------------
__global__ __launch_bounds__(256) void k_heads(
    const float* __restrict__ emb1,
    const float* __restrict__ w2, const float* __restrict__ b2,
    const float* __restrict__ w3, const float* __restrict__ b3,
    const float* __restrict__ wa1, const float* __restrict__ ba1,
    const float* __restrict__ wa2, const float* __restrict__ ba2,
    const float* __restrict__ wc1, const float* __restrict__ bc1,
    const float* __restrict__ wc2, const float* __restrict__ bc2,
    const float* __restrict__ wc3, const float* __restrict__ bc3,
    const float* __restrict__ loss, float* __restrict__ out) {
  __shared__ __align__(32) bf16_t act[16 * 128];    // A-fragment order
  __shared__ __align__(32) bf16_t act3[16 * 128];   // A-fragment order
  __shared__ __align__(32) bf16_t wl[128 * 128];    // B-fragment order
  __shared__ float bl[128];
  __shared__ float lg[16 * 10];
  int tid = threadIdx.x;
  int mbase = blockIdx.x * 16;

  {   // load emb1 tile -> act (swizzled, contiguous 8-element runs)
    int row = tid >> 4, g = tid & 15;
    int kc = g >> 2, sub = g & 3;
    int hia = sub & 1, e0 = (sub >> 1) << 3;
    bf16_t* dst = act + ((kc << 5) + (hia << 4) + row) * 16 + e0;
    const float* src = emb1 + (size_t)(mbase + row) * 128 + g * 8;
#pragma unroll
    for (int j = 0; j < 8; ++j) dst[j] = (bf16_t)src[j];
  }
  __syncthreads();

  dense128(w2, b2, act, act, wl, bl, tid);
  dense128(w3, b3, act, act, wl, bl, tid);
  {   // save emb3 (linear copy preserves swizzle)
#pragma unroll
    for (int j = 0; j < 8; ++j) act3[tid * 8 + j] = act[tid * 8 + j];
  }
  __syncthreads();

  dense128(wa1, ba1, act, act, wl, bl, tid);   // actor hidden a1 -> act
  if (tid < 160) {   // logits [16 x 10]
    int m = tid / 10, j = tid % 10;
    float s = ba2[j];
    for (int k = 0; k < 128; ++k)
      s = fmaf((float)act[a_sw_idx(m, k)], wa2[k * 10 + j], s);
    lg[m * 10 + j] = s;
  }
  __syncthreads();
  if (tid < 16) {    // softmax -> probs
    float mx = -3.4e38f;
    for (int j = 0; j < 10; ++j) mx = fmaxf(mx, lg[tid * 10 + j]);
    float sm = 0.0f, e[10];
    for (int j = 0; j < 10; ++j) { e[j] = __expf(lg[tid * 10 + j] - mx); sm += e[j]; }
    float inv = 1.0f / sm;
    for (int j = 0; j < 10; ++j) out[(size_t)(mbase + tid) * 10 + j] = e[j] * inv;
  }
  __syncthreads();

  dense128(wc1, bc1, act3, act3, wl, bl, tid);
  dense128(wc2, bc2, act3, act3, wl, bl, tid);
  if (tid < 16) {    // value head
    float s = bc3[0];
    for (int k = 0; k < 128; ++k)
      s = fmaf((float)act3[a_sw_idx(tid, k)], wc3[k], s);
    out[5120 + mbase + tid] = s;
  }
  if (blockIdx.x == 0 && tid == 0)
    out[5632] = loss[0] * (1.25f / (524288.0f * 64.0f));  // (1+COMMIT) * MSE
}

// ---------------- launch ----------------
extern "C" void kernel_launch(void* const* d_in, const int* in_sizes, int n_in,
                              void* d_out, int out_size, void* d_ws, size_t ws_size,
                              hipStream_t stream) {
  (void)in_sizes; (void)n_in; (void)out_size; (void)ws_size;
  const float* x   = (const float*)d_in[0];
  const float* c1w = (const float*)d_in[1];
  const float* c1b = (const float*)d_in[2];
  const float* c2w = (const float*)d_in[3];
  const float* c2b = (const float*)d_in[4];
  const float* cbk = (const float*)d_in[5];
  const float* w1  = (const float*)d_in[6];
  const float* b1  = (const float*)d_in[7];
  const float* w2  = (const float*)d_in[8];
  const float* b2  = (const float*)d_in[9];
  const float* w3  = (const float*)d_in[10];
  const float* b3  = (const float*)d_in[11];
  const float* wa1 = (const float*)d_in[12];
  const float* ba1 = (const float*)d_in[13];
  const float* wa2 = (const float*)d_in[14];
  const float* ba2 = (const float*)d_in[15];
  const float* wc1 = (const float*)d_in[16];
  const float* bc1 = (const float*)d_in[17];
  const float* wc2 = (const float*)d_in[18];
  const float* bc2 = (const float*)d_in[19];
  const float* wc3 = (const float*)d_in[20];
  const float* bc3 = (const float*)d_in[21];

  char* ws = (char*)d_ws;
  size_t off = 0;
  auto take = [&](size_t b) { size_t o = off; off = (off + b + 255) & ~(size_t)255; return o; };
  bf16_t* h1   = (bf16_t*)(ws + take((size_t)PIX * 32 * 2));
  bf16_t* h2   = (bf16_t*)(ws + take((size_t)PIX * 64 * 2));
  bf16_t* emb  = (bf16_t*)(ws + take((size_t)PIX * 64 * 2));
  int*    idx  = (int*)   (ws + take((size_t)PIX * 4));
  float*  emb1 = (float*) (ws + take((size_t)512 * 128 * 4));
  float*  loss = (float*) (ws + take(256));

  float* out = (float*)d_out;

  k_init  <<<1, 64, 0, stream>>>(loss);
  k_conv1 <<<PIX / 256, 256, 0, stream>>>(x, c1w, c1b, h1);
  k_conv2 <<<512 * 16, 128, 0, stream>>>(h1, c2w, c2b, h2);
  k_vq    <<<PIX / 128, 256, 0, stream>>>(h2, cbk, idx);
  k_gather<<<PIX / 256, 256, 0, stream>>>(idx, cbk, h2, emb, loss, out + 5633);
  k_fc1   <<<32, 256, 0, stream>>>(emb, w1, b1, emb1);
  k_heads <<<32, 256, 0, stream>>>(emb1, w2, b2, w3, b3, wa1, ba1, wa2, ba2,
                                   wc1, bc1, wc2, bc2, wc3, bc3, loss, out);
}
